// Hgru2d_60997125538134
// MI455X (gfx1250) — matrix operand order repaired and checked by reading, not compiled, not run on hardware
//
#include <hip/hip_runtime.h>
#include <hip/hip_bf16.h>
#include <stdint.h>

// Problem dims
#define HH 64
#define WW 64
#define BB 8
#define DD 512
#define MM (HH * WW * BB)   // 32768 tokens
#define ND2 (2 * DD)        // 1024

typedef __attribute__((ext_vector_type(16))) _Float16 v16h;
typedef __attribute__((ext_vector_type(8)))  float    v8f;
typedef __attribute__((ext_vector_type(8)))  _Float16 h8;

// GEMM epilogue modes
#define EPI_SILU_F32 0
#define EPI_SILU_F16 1
#define EPI_SIG_F32  2
#define EPI_ID_F32   3

__device__ __forceinline__ float fast_sigmoid(float v) {
    return __builtin_amdgcn_rcpf(1.f + __expf(-v));   // v_rcp_f32 path
}

// ---------------------------------------------------------------- converts
__global__ __launch_bounds__(256) void k_f32_to_f16(const float* __restrict__ src,
                                                    _Float16* __restrict__ dst, int n) {
    int i = blockIdx.x * blockDim.x + threadIdx.x;
    int stride = gridDim.x * blockDim.x;
    for (; i < n; i += stride) dst[i] = (_Float16)src[i];
}

// ---------------------------------------------------------------- WMMA GEMM
// C[M,N] = epilogue(A[M,K] @ B[K,N] + bias[N])
// Block tile 128x64, K step 32, double-buffered LDS. 8 waves; each wave owns
// a 32x32 output patch (2x2 subtiles -> 4 WMMAs, 2 A-frags x 2 B-frags).
// A tile: CDNA5 async global->LDS copy (ASYNCcnt) of the NEXT tile overlapped
// with WMMA on the current tile; one barrier per K-step. B tile staged
// transposed so fragment gathers are contiguous ds_load_b128s.
__global__ __launch_bounds__(256) void k_gemm_wmma(
    const _Float16* __restrict__ A, const _Float16* __restrict__ Bm,
    const float* __restrict__ bias,
    float* __restrict__ outF, _Float16* __restrict__ outH,
    int N, int K, int mode)
{
    __shared__ alignas(16) _Float16 sA [2][128][40];  // [buf][m][k] (+pad)
    __shared__ alignas(16) _Float16 sBt[2][ 64][40];  // [buf][n][k] (+pad)

    const int t   = threadIdx.x;
    const int wv  = t >> 5;
    const int ln  = t & 31;
    const int hl  = ln & 15;        // lane within half-wave
    const int hiL = ln >> 4;        // which half-wave
    const int n0  = (wv & 1) * 32;  // wave's N offset (2 subtiles)
    const int mB  = (wv >> 1) * 32; // wave's M offset (2 subtiles)
    const long rowBlk = (long)blockIdx.x * 128;
    const int  nb     = blockIdx.y * 64;

    // cooperative tile-load coordinates
    const int rA0 = t >> 1, cA0 = (t & 1) * 16;  // A: 128 rows x 32 k, 32B/thread
    const int rB  = t >> 3, cB  = (t & 7) * 8;   // B: 32 k x 64 n, 16B/thread

    // --- tile movers -------------------------------------------------------
    auto issueA = [&](int kb, int buf) {
        const _Float16* aSrc = A + (rowBlk + rA0) * K + kb + cA0;
#if defined(__gfx1250__)
        uint64_t ga = (uint64_t)(uintptr_t)aSrc;
        uint32_t la = (uint32_t)(uintptr_t)(&sA[buf][rA0][cA0]);
        asm volatile("global_load_async_to_lds_b128 %0, %1, off"
                     :: "v"(la), "v"(ga) : "memory");
        asm volatile("global_load_async_to_lds_b128 %0, %1, off"
                     :: "v"(la + 16u), "v"(ga + 16u) : "memory");
#else
        *(h8*)(&sA[buf][rA0][cA0])     = *(const h8*)aSrc;
        *(h8*)(&sA[buf][rA0][cA0 + 8]) = *(const h8*)(aSrc + 8);
#endif
    };
    auto loadB = [&](int kb) -> h8 {
        return *(const h8*)(Bm + (long)(kb + rB) * N + nb + cB);
    };
    auto scatterB = [&](h8 bv, int buf) {
        #pragma unroll
        for (int j = 0; j < 8; ++j) sBt[buf][cB + j][rB] = bv[j];
    };

    v8f acc00 = {}, acc01 = {}, acc10 = {}, acc11 = {};

    // --- prologue: stage tile 0 into buffer 0 ------------------------------
    issueA(0, 0);
    scatterB(loadB(0), 0);
#if defined(__gfx1250__)
    asm volatile("s_wait_asynccnt 0x0" ::: "memory");
#endif
    __syncthreads();

    // --- main loop ---------------------------------------------------------
    for (int kb = 0; kb < K; kb += 32) {
        const int  cur     = (kb >> 5) & 1;
        const int  nxt     = cur ^ 1;
        const bool hasNext = (kb + 32 < K);

        h8 bvN;
        if (hasNext) {
            issueA(kb + 32, nxt);          // async copy overlaps WMMA below
            bvN = loadB(kb + 32);
            if (kb + 64 < K) {             // gfx1250 global_prefetch_b8
                __builtin_prefetch(A + (rowBlk + rA0) * K + kb + 64 + cA0, 0, 3);
                __builtin_prefetch(Bm + (long)(kb + 64 + rB) * N + nb + cB, 0, 3);
            }
        }

        // B fragments: lane holds column, K = (j>=8?16:0) + hiL*8 + (j&7) (matches A)
        v16h bf0, bf1;
        {
            const _Float16* bp0 = &sBt[cur][n0 + hl][hiL * 8];
            const _Float16* bp1 = &sBt[cur][n0 + 16 + hl][hiL * 8];
            #pragma unroll
            for (int j = 0; j < 16; ++j) { bf0[j] = bp0[((j & 8) << 1) + (j & 7)]; bf1[j] = bp1[((j & 8) << 1) + (j & 7)]; }
        }
        // A fragments: lane holds row; VGPR v packs K pairs
        // k0 = (v>=4?16:0) + hiL*8 + (v&3)*2  (ISA 16-bit A 16x32 layout)
        v16h a0, a1;
        {
            const _Float16* ap0 = &sA[cur][mB + hl][0];
            const _Float16* ap1 = &sA[cur][mB + 16 + hl][0];
            #pragma unroll
            for (int v = 0; v < 8; ++v) {
                const int k0 = ((v & 4) << 2) + hiL * 8 + (v & 3) * 2;
                a0[2 * v]     = ap0[k0];
                a0[2 * v + 1] = ap0[k0 + 1];
                a1[2 * v]     = ap1[k0];
                a1[2 * v + 1] = ap1[k0 + 1];
            }
        }
        acc00 = __builtin_amdgcn_wmma_f32_16x16x32_f16(false, a0, false, bf0,
                                                       (short)0, acc00, false, false);
        acc01 = __builtin_amdgcn_wmma_f32_16x16x32_f16(false, a0, false, bf1,
                                                       (short)0, acc01, false, false);
        acc10 = __builtin_amdgcn_wmma_f32_16x16x32_f16(false, a1, false, bf0,
                                                       (short)0, acc10, false, false);
        acc11 = __builtin_amdgcn_wmma_f32_16x16x32_f16(false, a1, false, bf1,
                                                       (short)0, acc11, false, false);

        if (hasNext) {
            scatterB(bvN, nxt);
#if defined(__gfx1250__)
            asm volatile("s_wait_asynccnt 0x0" ::: "memory");
#endif
            __syncthreads();               // publishes nxt, protects cur
        }
    }

    // Epilogue. D layout: element v of lane = row (v + hiL*8), col (hl).
    #pragma unroll
    for (int mi = 0; mi < 2; ++mi) {
        #pragma unroll
        for (int nj = 0; nj < 2; ++nj) {
            v8f acc = (mi == 0) ? (nj == 0 ? acc00 : acc01)
                                : (nj == 0 ? acc10 : acc11);
            const int  col = nb + n0 + nj * 16 + hl;
            const float bsv = bias[col];
            const long r0 = rowBlk + mB + mi * 16 + hiL * 8;
            #pragma unroll
            for (int v = 0; v < 8; ++v) {
                float val = acc[v] + bsv;
                long  o   = (r0 + v) * N + col;
                if (mode == EPI_SILU_F32) {
                    outF[o] = val * fast_sigmoid(val);
                } else if (mode == EPI_SILU_F16) {
                    outH[o] = (_Float16)(val * fast_sigmoid(val));
                } else if (mode == EPI_SIG_F32) {
                    outF[o] = fast_sigmoid(val);
                } else {
                    outF[o] = val;
                }
            }
        }
    }
}

// ---------------------------------------------------------------- scans
// h-scan: one thread per (w,b,d) lane; 64 sequential steps along h.
__global__ __launch_bounds__(256) void k_scan_h(
    const float* __restrict__ lam, const float* __restrict__ is2,
    const float* __restrict__ theta,
    float* __restrict__ hr, float* __restrict__ hi)
{
    const int t = blockIdx.x * blockDim.x + threadIdx.x;   // 0..W*B*D-1
    const int d = t & (DD - 1);
    const float th = theta[d];
    const float c = __cosf(th), s = __sinf(th);
    const long S = (long)WW * BB * DD;                     // h stride
    float ar = 0.f, ai = 0.f;
    for (int h = 0; h < HH; ++h) {
        const long idx = (long)h * S + t;
        const float l  = lam[idx];
        const float u0 = is2[2 * idx], u1 = is2[2 * idx + 1];
        const float om = 1.f - l;
        const float gr = l * c, gi = l * s;
        const float nr = gr * ar - gi * ai + om * u0;
        const float ni = gr * ai + gi * ar + om * u1;
        ar = nr; ai = ni;
        hr[idx] = nr;
        hi[idx] = ni;
    }
}

// w-scan: one thread per (h,b,d) lane; accumulates into hr/hi.
__global__ __launch_bounds__(256) void k_scan_w(
    const float* __restrict__ lam, const float* __restrict__ is2,
    const float* __restrict__ theta,
    float* __restrict__ hr, float* __restrict__ hi)
{
    const int t  = blockIdx.x * blockDim.x + threadIdx.x;  // 0..H*B*D-1
    const int d  = t & (DD - 1);
    const int h  = t >> 12;                                // / (B*D)
    const int bd = t & (BB * DD - 1);
    const float th = theta[d];
    const float c = __cosf(th), s = __sinf(th);
    float ar = 0.f, ai = 0.f;
    for (int w = 0; w < WW; ++w) {
        const long idx = (long)h * (WW * BB * DD) + (long)w * (BB * DD) + bd;
        const float l  = lam[idx];
        const float u0 = is2[2 * idx], u1 = is2[2 * idx + 1];
        const float om = 1.f - l;
        const float gr = l * c, gi = l * s;
        const float nr = gr * ar - gi * ai + om * u0;
        const float ni = gr * ai + gi * ar + om * u1;
        ar = nr; ai = ni;
        hr[idx] += nr;
        hi[idx] += ni;
    }
}

// ---------------------------------------------------------------- LN * gate
__global__ __launch_bounds__(256) void k_ln_gate(
    const float* __restrict__ hr, const float* __restrict__ hi,
    const _Float16* __restrict__ gate,
    const float* __restrict__ g, const float* __restrict__ b,
    _Float16* __restrict__ fg)
{
    __shared__ float red[16];
    const long tok = blockIdx.x;
    const float* pr = hr + tok * DD;
    const float* pi = hi + tok * DD;
    float vals[4];
    float sum = 0.f, sq = 0.f;
    #pragma unroll
    for (int i = 0; i < 4; ++i) {
        const int j = threadIdx.x + i * 256;
        const float v = (j < DD) ? pr[j] : pi[j - DD];
        vals[i] = v; sum += v; sq += v * v;
    }
    #pragma unroll
    for (int off = 16; off > 0; off >>= 1) {           // wave32 reduce
        sum += __shfl_down(sum, off, 32);
        sq  += __shfl_down(sq,  off, 32);
    }
    const int wv = threadIdx.x >> 5, ln = threadIdx.x & 31;
    if (ln == 0) { red[wv] = sum; red[8 + wv] = sq; }
    __syncthreads();
    if (threadIdx.x == 0) {
        float s0 = 0.f, q0 = 0.f;
        for (int i = 0; i < 8; ++i) { s0 += red[i]; q0 += red[8 + i]; }
        const float mu  = s0 / (float)ND2;
        const float var = q0 / (float)ND2 - mu * mu;
        red[0] = mu;
        red[8] = rsqrtf(var + 1e-5f);
    }
    __syncthreads();
    const float mu = red[0], rstd = red[8];
    #pragma unroll
    for (int i = 0; i < 4; ++i) {
        const int j = threadIdx.x + i * 256;
        const float f  = (vals[i] - mu) * rstd * g[j] + b[j];
        const float gt = (float)gate[tok * ND2 + j];
        fg[tok * ND2 + j] = (_Float16)(f * gt);
    }
}

// ---------------------------------------------------------------- launch
extern "C" void kernel_launch(void* const* d_in, const int* in_sizes, int n_in,
                              void* d_out, int out_size, void* d_ws, size_t ws_size,
                              hipStream_t stream) {
    (void)in_sizes; (void)n_in; (void)out_size; (void)ws_size;
    const float* x     = (const float*)d_in[0];
    const float* Wi    = (const float*)d_in[1];
    const float* bi    = (const float*)d_in[2];
    const float* Wl    = (const float*)d_in[3];
    const float* bl    = (const float*)d_in[4];
    const float* theta = (const float*)d_in[5];
    const float* Wg    = (const float*)d_in[6];
    const float* bg    = (const float*)d_in[7];
    const float* Wo    = (const float*)d_in[8];
    const float* bo    = (const float*)d_in[9];
    const float* lng   = (const float*)d_in[10];
    const float* lnb   = (const float*)d_in[11];
    float* out = (float*)d_out;

    char* ws = (char*)d_ws;
    size_t off = 0;
    auto carve = [&](size_t bytes) -> void* {
        off = (off + 255) & ~(size_t)255;
        void* p = ws + off;
        off += bytes;
        return p;
    };
    _Float16* xh    = (_Float16*)carve((size_t)MM * DD * 2);
    _Float16* WiH   = (_Float16*)carve((size_t)DD * ND2 * 2);
    _Float16* WgH   = (_Float16*)carve((size_t)DD * ND2 * 2);
    _Float16* WlH   = (_Float16*)carve((size_t)DD * DD * 2);
    _Float16* WoH   = (_Float16*)carve((size_t)ND2 * DD * 2);
    float*    ist   = (float*)carve((size_t)MM * ND2 * 4);   // silu(x@Wi+bi)
    _Float16* gateH = (_Float16*)carve((size_t)MM * ND2 * 2);
    float*    lamB  = (float*)carve((size_t)MM * DD * 4);
    float*    hrB   = (float*)carve((size_t)MM * DD * 4);
    float*    hiB   = (float*)carve((size_t)MM * DD * 4);
    _Float16* fgB   = (_Float16*)carve((size_t)MM * ND2 * 2);

    const dim3 blk(256);

    // f32 -> f16 conversions
    k_f32_to_f16<<<1024, blk, 0, stream>>>(x,  xh,  MM * DD);
    k_f32_to_f16<<<512,  blk, 0, stream>>>(Wi, WiH, DD * ND2);
    k_f32_to_f16<<<512,  blk, 0, stream>>>(Wg, WgH, DD * ND2);
    k_f32_to_f16<<<256,  blk, 0, stream>>>(Wl, WlH, DD * DD);
    k_f32_to_f16<<<512,  blk, 0, stream>>>(Wo, WoH, ND2 * DD);

    // input GEMMs (WMMA f16 -> f32 acc, fused activations)
    const dim3 gN1(MM / 128, ND2 / 64);
    const dim3 gN0(MM / 128, DD / 64);
    k_gemm_wmma<<<gN1, blk, 0, stream>>>(xh, WiH, bi, ist, (_Float16*)nullptr,
                                         ND2, DD, EPI_SILU_F32);
    k_gemm_wmma<<<gN1, blk, 0, stream>>>(xh, WgH, bg, (float*)nullptr, gateH,
                                         ND2, DD, EPI_SILU_F16);
    k_gemm_wmma<<<gN0, blk, 0, stream>>>(xh, WlH, bl, lamB, (_Float16*)nullptr,
                                         DD, DD, EPI_SIG_F32);

    // directional complex scans
    k_scan_h<<<(WW * BB * DD) / 256, blk, 0, stream>>>(lamB, ist, theta, hrB, hiB);
    k_scan_w<<<(HH * BB * DD) / 256, blk, 0, stream>>>(lamB, ist, theta, hrB, hiB);

    // layernorm * gate -> f16
    k_ln_gate<<<MM, blk, 0, stream>>>(hrB, hiB, gateH, lng, lnb, fgB);

    // output GEMM -> d_out (f32)
    k_gemm_wmma<<<gN0, blk, 0, stream>>>(fgB, WoH, bo, out, (_Float16*)nullptr,
                                         DD, ND2, EPI_ID_F32);
}
